// RelativePositionAttention_74577812128322
// MI455X (gfx1250) — matrix operand order repaired and checked by reading, not compiled, not run on hardware
//
#include <hip/hip_runtime.h>
#include <hip/hip_bf16.h>
#include <cstdint>
#include <cstddef>

// ---------------------------------------------------------------------------
// Types for CDNA5 WMMA (wave32): D(f32 16x16) = A(bf16 16x32) x B(bf16 32x16) + C
// ---------------------------------------------------------------------------
typedef __bf16 bf16_t;
typedef __attribute__((ext_vector_type(16))) __bf16 v16bf;
typedef __attribute__((ext_vector_type(8)))  float  v8f;

#define E_DIM   768
#define H_HEADS 12
#define D_HEAD  64
#define B_BATCH 4
#define N_SEQ   1024
#define M_ROWS  (B_BATCH * N_SEQ)   // 4096
#define F_DIM   (4 * E_DIM)         // 3072
#define LN_EPS  1e-5f
#define SM_SCALE 0.125f             // 1/sqrt(64)

__device__ __forceinline__ bf16_t f2bf(float f) { return (bf16_t)f; }

__device__ __forceinline__ v8f wmma_bf16(v16bf a, v16bf b, v8f c) {
  // (neg_a, A, neg_b, B, c_mod, C, reuse_a, reuse_b)
  return __builtin_amdgcn_wmma_f32_16x16x32_bf16(false, a, false, b, (short)0, c,
                                                 false, false);
}

// XOR-lane shuffle via ds_swizzle (group-of-32 mode: offset = xor<<10 | and=0x1f)
template <int MASK>
__device__ __forceinline__ float swz_xor(float v) {
  return __int_as_float(
      __builtin_amdgcn_ds_swizzle(__float_as_int(v), (MASK << 10) | 0x1f));
}
__device__ __forceinline__ float redmax16(float v) {
  v = fmaxf(v, swz_xor<1>(v));
  v = fmaxf(v, swz_xor<2>(v));
  v = fmaxf(v, swz_xor<4>(v));
  v = fmaxf(v, swz_xor<8>(v));
  return v;
}
__device__ __forceinline__ float redsum16(float v) {
  v += swz_xor<1>(v);
  v += swz_xor<2>(v);
  v += swz_xor<4>(v);
  v += swz_xor<8>(v);
  return v;
}

// ---------------------------------------------------------------------------
// fp32 (RxC, row-major) -> bf16 transposed (CxR, row-major), LDS-tiled.
// Block (32,8); grid (C/32, R/32). R,C multiples of 32.
// ---------------------------------------------------------------------------
__global__ void cvt_transpose_kernel(const float* __restrict__ in,
                                     bf16_t* __restrict__ out, int R, int C) {
  __shared__ bf16_t tile[32][33];
  int c0 = blockIdx.x * 32;
  int r0 = blockIdx.y * 32;
  int tx = threadIdx.x;   // 0..31
  int ty = threadIdx.y;   // 0..7
#pragma unroll
  for (int j = 0; j < 4; ++j) {
    int r = r0 + ty + j * 8;
    tile[ty + j * 8][tx] = f2bf(in[(size_t)r * C + c0 + tx]);
  }
  __syncthreads();
#pragma unroll
  for (int j = 0; j < 4; ++j) {
    int c = c0 + ty + j * 8;
    out[(size_t)c * R + r0 + tx] = tile[tx][ty + j * 8];
  }
}

// ---------------------------------------------------------------------------
// Row LayerNorm over E=768: one block (256 thr) per row, 3 elems/thread.
// ---------------------------------------------------------------------------
template <bool WRITE_F32>
__global__ void layernorm_kernel(const float* __restrict__ x,
                                 const float* __restrict__ g,
                                 const float* __restrict__ beta,
                                 float* __restrict__ out_f32,
                                 bf16_t* __restrict__ out_bf16) {
  __shared__ float s_sum[256];
  __shared__ float s_sq[256];
  int row = blockIdx.x;
  const float* xr = x + (size_t)row * E_DIM;
  float lx[3];
  float s = 0.f, s2 = 0.f;
#pragma unroll
  for (int j = 0; j < 3; ++j) {
    float v = xr[threadIdx.x + j * 256];
    lx[j] = v; s += v; s2 += v * v;
  }
  s_sum[threadIdx.x] = s; s_sq[threadIdx.x] = s2;
  __syncthreads();
  for (int off = 128; off > 0; off >>= 1) {
    if (threadIdx.x < off) {
      s_sum[threadIdx.x] += s_sum[threadIdx.x + off];
      s_sq[threadIdx.x]  += s_sq[threadIdx.x + off];
    }
    __syncthreads();
  }
  float mean = s_sum[0] * (1.0f / E_DIM);
  float var  = s_sq[0] * (1.0f / E_DIM) - mean * mean;
  float inv  = rsqrtf(var + LN_EPS);
#pragma unroll
  for (int j = 0; j < 3; ++j) {
    int c = threadIdx.x + j * 256;
    float v = (lx[j] - mean) * inv * g[c] + beta[c];
    if (WRITE_F32) out_f32[(size_t)row * E_DIM + c] = v;
    out_bf16[(size_t)row * E_DIM + c] = f2bf(v);
  }
}

// ---------------------------------------------------------------------------
// WMMA GEMM: C(MxN) = A_bf16(MxK, row-major) x Wt_bf16(NxK, row-major) + bias
// Wave tile 32x64: 2 A-frags x 4 B-frags -> 8 WMMAs per K-step (32).
// ACT: exact GELU ; RES: fp32 residual ; OUTF32/OUTBF16 outputs;
// OUTVT: bf16 output stored transposed per (batch,head): [b][h][d][n].
// ---------------------------------------------------------------------------
template <int ACT, int RES, int OUTF32, int OUTBF16, int OUTVT>
__global__ void gemm_bias_kernel(const bf16_t* __restrict__ A,
                                 const bf16_t* __restrict__ Wt,
                                 const float* __restrict__ bias,
                                 const float* __restrict__ resid,
                                 float* __restrict__ outF,
                                 bf16_t* __restrict__ outB,
                                 int M, int K, int Nc) {
  int lane = threadIdx.x & 31;
  int wave = threadIdx.x >> 5;
  int tiles_m = M >> 5;                    // 32-row tiles
  int tile = blockIdx.x * 8 + wave;
  int tm = tile % tiles_m;
  int tn = tile / tiles_m;
  int half = lane >> 4;
  int l16  = lane & 15;

  const bf16_t* A0 = A + (size_t)(tm * 32 + l16) * K;
  const bf16_t* A1 = A0 + (size_t)16 * K;
  v8f acc[2][4] = {};

  for (int k0 = 0; k0 < K; k0 += 32) {
    v16bf a0, a1;
#pragma unroll
    for (int i = 0; i < 16; ++i) {
      int kk = k0 + half * 16 + i;
      a0[i] = A0[kk];
      a1[i] = A1[kk];
    }
#pragma unroll
    for (int j = 0; j < 4; ++j) {
      const bf16_t* Wr = Wt + (size_t)(tn * 64 + j * 16 + l16) * K + k0 + half * 16;
      v16bf bfrag;
#pragma unroll
      for (int i = 0; i < 16; ++i) bfrag[i] = Wr[i];
      acc[0][j] = wmma_bf16(a0, bfrag, acc[0][j]);
      acc[1][j] = wmma_bf16(a1, bfrag, acc[1][j]);
    }
  }

#pragma unroll
  for (int u = 0; u < 2; ++u)
#pragma unroll
    for (int j = 0; j < 4; ++j) {
      int ncol = tn * 64 + j * 16 + l16;
      float bv = bias[ncol];
#pragma unroll
      for (int r = 0; r < 8; ++r) {
        int row = tm * 32 + u * 16 + half * 8 + r;
        float v = acc[u][j][r] + bv;
        if (ACT) v = 0.5f * v * (1.0f + erff(v * 0.70710678118654752f));
        if (RES) v += resid[(size_t)row * Nc + ncol];
        if (OUTF32) outF[(size_t)row * Nc + ncol] = v;
        if (OUTBF16) {
          if (OUTVT) {
            int bb = row >> 10, n = row & (N_SEQ - 1);
            int hh = ncol >> 6, d = ncol & 63;
            outB[(((size_t)bb * H_HEADS + hh) * D_HEAD + d) * N_SEQ + n] = f2bf(v);
          } else {
            outB[(size_t)row * Nc + ncol] = f2bf(v);
          }
        }
      }
    }
}

// ---------------------------------------------------------------------------
// Flash attention with relative-position bias.
// Block = 256 thr (8 waves); wave owns 16 query rows; 32-key tiles; head fixed.
// Q is pre-scaled by 1/sqrt(D) (exact). bias*scale = qbias[n] - kbias[m] with
// kbias[m] = (cx[m]*wr0 + cy[m]*wr1)*scale cached in LDS for the whole block.
// V is consumed transposed: Vt[b][h][d][n].
// ---------------------------------------------------------------------------
__global__ void flash_attn_kernel(const bf16_t* __restrict__ Q,
                                  const bf16_t* __restrict__ Kmat,
                                  const bf16_t* __restrict__ Vt,
                                  const float* __restrict__ coords,
                                  const float* __restrict__ w_rel,
                                  const float* __restrict__ b_rel,
                                  bf16_t* __restrict__ Out) {
  __shared__ float  kbias_s[N_SEQ];        // 4 KB
  __shared__ bf16_t lds_p[8][16 * 32];     // 8 KB: per-wave P tile

  int lane = threadIdx.x & 31;
  int wave = threadIdx.x >> 5;
  const int qblks = N_SEQ / 128;           // 8
  int bh   = blockIdx.x / qblks;
  int qblk = blockIdx.x % qblks;
  int b = bh / H_HEADS;
  int h = bh % H_HEADS;
  int half = lane >> 4;
  int l16  = lane & 15;
  int qrow0 = qblk * 128 + wave * 16;
  const size_t base = (size_t)b * N_SEQ;

  float wr0 = w_rel[h];
  float wr1 = w_rel[H_HEADS + h];
  float brs = b_rel[h] * SM_SCALE;

  // key-side bias table for the whole block
  for (int i = threadIdx.x; i < N_SEQ; i += 256) {
    float cx = coords[(base + i) * 2 + 0];
    float cy = coords[(base + i) * 2 + 1];
    kbias_s[i] = (cx * wr0 + cy * wr1) * SM_SCALE;
  }
  __syncthreads();

  // Resident Q A-fragments (pre-scaled by 1/sqrt(D)): 16x64 = two K=32 frags.
  const bf16_t* qrow = Q + (base + qrow0 + l16) * E_DIM + h * D_HEAD;
  v16bf qa0, qa1;
#pragma unroll
  for (int i = 0; i < 16; ++i) {
    int d = half * 16 + i;
    qa0[i] = f2bf((float)qrow[d] * SM_SCALE);
    qa1[i] = f2bf((float)qrow[32 + d] * SM_SCALE);
  }

  float mrow[8], lrow[8], qbias[8];
#pragma unroll
  for (int r = 0; r < 8; ++r) {
    mrow[r] = -1e30f;
    lrow[r] = 0.0f;
    qbias[r] = kbias_s[qrow0 + half * 8 + r] + brs;
  }
  v8f oacc[4] = {};

  for (int m0 = 0; m0 < N_SEQ; m0 += 32) {
    // ---- scores: two 16x16 tiles over 32 keys, K=64 in two WMMAs each ----
    v8f s0 = {}, s1 = {};
    {
      const bf16_t* k0p = Kmat + (base + m0 + l16) * E_DIM + h * D_HEAD + half * 16;
      const bf16_t* k1p = Kmat + (base + m0 + 16 + l16) * E_DIM + h * D_HEAD + half * 16;
      v16bf kb;
#pragma unroll
      for (int i = 0; i < 16; ++i) kb[i] = k0p[i];
      s0 = wmma_bf16(qa0, kb, s0);
#pragma unroll
      for (int i = 0; i < 16; ++i) kb[i] = k0p[32 + i];
      s0 = wmma_bf16(qa1, kb, s0);
#pragma unroll
      for (int i = 0; i < 16; ++i) kb[i] = k1p[i];
      s1 = wmma_bf16(qa0, kb, s1);
#pragma unroll
      for (int i = 0; i < 16; ++i) kb[i] = k1p[32 + i];
      s1 = wmma_bf16(qa1, kb, s1);
    }

    float kb0 = kbias_s[m0 + l16];
    float kb1 = kbias_s[m0 + 16 + l16];
#pragma unroll
    for (int r = 0; r < 8; ++r) {
      float sv0 = s0[r] + qbias[r] - kb0;
      float sv1 = s1[r] + qbias[r] - kb1;
      float t = redmax16(fmaxf(sv0, sv1));
      float newm = fmaxf(mrow[r], t);
      float corr = __expf(mrow[r] - newm);
      float e0 = __expf(sv0 - newm);
      float e1 = __expf(sv1 - newm);
      float rs = redsum16(e0 + e1);
      lrow[r] = lrow[r] * corr + rs;
      mrow[r] = newm;
      oacc[0][r] *= corr; oacc[1][r] *= corr;
      oacc[2][r] *= corr; oacc[3][r] *= corr;
      int rowl = half * 8 + r;
      lds_p[wave][rowl * 32 + l16]      = f2bf(e0);
      lds_p[wave][rowl * 32 + 16 + l16] = f2bf(e1);
    }

    // ---- P (C-layout in LDS) -> A-fragment, then O += P x V (K=32) ----
    v16bf pa;
#pragma unroll
    for (int i = 0; i < 16; ++i) {
      int kk = half * 16 + i;
      pa[i] = lds_p[wave][l16 * 32 + kk];
    }
#pragma unroll
    for (int j = 0; j < 4; ++j) {
      // Vt[b][h][d][n]: contiguous along keys
      const bf16_t* vp = Vt + (((size_t)b * H_HEADS + h) * D_HEAD + j * 16 + l16) * N_SEQ
                       + m0 + half * 16;
      v16bf vb;
#pragma unroll
      for (int i = 0; i < 16; ++i) vb[i] = vp[i];
      oacc[j] = wmma_bf16(pa, vb, oacc[j]);
    }
  }

  // ---- normalize and store (bf16 row-major for the out-proj GEMM) ----
#pragma unroll
  for (int j = 0; j < 4; ++j)
#pragma unroll
    for (int r = 0; r < 8; ++r) {
      int row = qrow0 + half * 8 + r;
      Out[(base + row) * E_DIM + h * D_HEAD + j * 16 + l16] =
          f2bf(oacc[j][r] / lrow[r]);
    }
}

// ---------------------------------------------------------------------------
// Host-side orchestration
// ---------------------------------------------------------------------------
extern "C" void kernel_launch(void* const* d_in, const int* in_sizes, int n_in,
                              void* d_out, int out_size, void* d_ws, size_t ws_size,
                              hipStream_t stream) {
  const float* x      = (const float*)d_in[0];
  const float* coords = (const float*)d_in[1];
  const float* wq = (const float*)d_in[2];   const float* bq = (const float*)d_in[3];
  const float* wk = (const float*)d_in[4];   const float* bk = (const float*)d_in[5];
  const float* wv = (const float*)d_in[6];   const float* bv = (const float*)d_in[7];
  const float* wo = (const float*)d_in[8];   const float* bo = (const float*)d_in[9];
  const float* w_rel = (const float*)d_in[10]; const float* b_rel = (const float*)d_in[11];
  const float* g1 = (const float*)d_in[12];  const float* beta1 = (const float*)d_in[13];
  const float* g2 = (const float*)d_in[14];  const float* beta2 = (const float*)d_in[15];
  const float* w_ffn1 = (const float*)d_in[16]; const float* b_ffn1 = (const float*)d_in[17];
  const float* w_ffn2 = (const float*)d_in[18]; const float* b_ffn2 = (const float*)d_in[19];
  float* out = (float*)d_out;
  (void)in_sizes; (void)n_in; (void)out_size; (void)ws_size;

  char* ws = (char*)d_ws;
  size_t off = 0;
  auto carve = [&](size_t bytes) -> void* {
    void* p = ws + off;
    off += (bytes + 255) & ~(size_t)255;
    return p;
  };

  // transposed bf16 weights (N x K)
  bf16_t* wq_t = (bf16_t*)carve((size_t)E_DIM * E_DIM * 2);
  bf16_t* wk_t = (bf16_t*)carve((size_t)E_DIM * E_DIM * 2);
  bf16_t* wv_t = (bf16_t*)carve((size_t)E_DIM * E_DIM * 2);
  bf16_t* wo_t = (bf16_t*)carve((size_t)E_DIM * E_DIM * 2);
  bf16_t* w1_t = (bf16_t*)carve((size_t)E_DIM * F_DIM * 2);  // 3072 x 768
  bf16_t* w2_t = (bf16_t*)carve((size_t)F_DIM * E_DIM * 2);  // 768 x 3072
  float*  xn_f = (float*) carve((size_t)M_ROWS * E_DIM * 4);
  bf16_t* xn_b = (bf16_t*)carve((size_t)M_ROWS * E_DIM * 2);
  bf16_t* q_b  = (bf16_t*)carve((size_t)M_ROWS * E_DIM * 2);
  bf16_t* k_b  = (bf16_t*)carve((size_t)M_ROWS * E_DIM * 2);
  bf16_t* vt_b = (bf16_t*)carve((size_t)M_ROWS * E_DIM * 2); // [b][h][d][n]
  bf16_t* at_b = (bf16_t*)carve((size_t)M_ROWS * E_DIM * 2);
  float*  x1_f = (float*) carve((size_t)M_ROWS * E_DIM * 4);
  bf16_t* h_b  = (bf16_t*)carve((size_t)M_ROWS * E_DIM * 2);
  bf16_t* f1_b = (bf16_t*)carve((size_t)M_ROWS * F_DIM * 2);

  // 1) weights -> bf16, transposed
  dim3 tb(32, 8);
  cvt_transpose_kernel<<<dim3(E_DIM / 32, E_DIM / 32), tb, 0, stream>>>(wq, wq_t, E_DIM, E_DIM);
  cvt_transpose_kernel<<<dim3(E_DIM / 32, E_DIM / 32), tb, 0, stream>>>(wk, wk_t, E_DIM, E_DIM);
  cvt_transpose_kernel<<<dim3(E_DIM / 32, E_DIM / 32), tb, 0, stream>>>(wv, wv_t, E_DIM, E_DIM);
  cvt_transpose_kernel<<<dim3(E_DIM / 32, E_DIM / 32), tb, 0, stream>>>(wo, wo_t, E_DIM, E_DIM);
  cvt_transpose_kernel<<<dim3(F_DIM / 32, E_DIM / 32), tb, 0, stream>>>(w_ffn1, w1_t, E_DIM, F_DIM);
  cvt_transpose_kernel<<<dim3(E_DIM / 32, F_DIM / 32), tb, 0, stream>>>(w_ffn2, w2_t, F_DIM, E_DIM);

  // 2) norm1
  layernorm_kernel<true><<<M_ROWS, 256, 0, stream>>>(x, g1, beta1, xn_f, xn_b);

  // 3) QKV projections (M=4096,K=768,N=768): (128*12)/8 = 192 blocks
  gemm_bias_kernel<0, 0, 0, 1, 0><<<192, 256, 0, stream>>>(
      xn_b, wq_t, bq, nullptr, nullptr, q_b, M_ROWS, E_DIM, E_DIM);
  gemm_bias_kernel<0, 0, 0, 1, 0><<<192, 256, 0, stream>>>(
      xn_b, wk_t, bk, nullptr, nullptr, k_b, M_ROWS, E_DIM, E_DIM);
  gemm_bias_kernel<0, 0, 0, 1, 1><<<192, 256, 0, stream>>>(
      xn_b, wv_t, bv, nullptr, nullptr, vt_b, M_ROWS, E_DIM, E_DIM);

  // 4) fused attention: B*H*(N/128) = 384 blocks of 256 threads
  flash_attn_kernel<<<B_BATCH * H_HEADS * (N_SEQ / 128), 256, 0, stream>>>(
      q_b, k_b, vt_b, coords, w_rel, b_rel, at_b);

  // 5) out-proj + residual with xn -> x1 (fp32)
  gemm_bias_kernel<0, 1, 1, 0, 0><<<192, 256, 0, stream>>>(
      at_b, wo_t, bo, xn_f, x1_f, nullptr, M_ROWS, E_DIM, E_DIM);

  // 6) norm2
  layernorm_kernel<false><<<M_ROWS, 256, 0, stream>>>(x1_f, g2, beta2, nullptr, h_b);

  // 7) FFN1 + exact GELU (M=4096,K=768,N=3072): (128*48)/8 = 768 blocks
  gemm_bias_kernel<1, 0, 0, 1, 0><<<768, 256, 0, stream>>>(
      h_b, w1_t, b_ffn1, nullptr, nullptr, f1_b, M_ROWS, E_DIM, F_DIM);

  // 8) FFN2 + residual with x1 -> final output (fp32)
  gemm_bias_kernel<0, 1, 1, 0, 0><<<192, 256, 0, stream>>>(
      f1_b, w2_t, b_ffn2, x1_f, out, nullptr, M_ROWS, F_DIM, E_DIM);
}
